// GroupedExperts_39436389712590
// MI455X (gfx1250) — matrix operand order repaired and checked by reading, not compiled, not run on hardware
//
#include <hip/hip_runtime.h>

#define E_ 8
#define H_ 1408
#define D_ 2048

#define BM 128
#define BN 64
#define BK 64
#define BKP 72   // +8 bf16 (16B) row padding to stagger LDS banks; row stride 144B (16B-aligned)

typedef __attribute__((ext_vector_type(8)))  __bf16 v8bf;
typedef __attribute__((ext_vector_type(16))) __bf16 v16bf;
typedef __attribute__((ext_vector_type(8)))  float  v8f;
typedef __attribute__((ext_vector_type(4)))  int    v4i;

typedef __attribute__((address_space(1))) v4i gv4i;   // global 128-bit chunk
typedef __attribute__((address_space(3))) v4i lv4i;   // LDS 128-bit chunk

#if __has_builtin(__builtin_amdgcn_global_load_async_to_lds_b128) && \
    __has_builtin(__builtin_amdgcn_s_wait_asynccnt)
#define USE_ASYNC 1
#else
#define USE_ASYNC 0
#endif

// 16-byte global -> LDS copy. Async (ASYNCcnt, no VGPR round trip) when available.
static __device__ __forceinline__ void cp16(const __bf16* g, __bf16* l) {
#if USE_ASYNC
    __builtin_amdgcn_global_load_async_to_lds_b128(
        (gv4i*)(uintptr_t)g,
        (lv4i*)(unsigned)(uintptr_t)l,   // low 32 bits of flat LDS ptr == LDS offset
        0, 0);
#else
    *(v8bf*)l = *(const v8bf*)g;
#endif
}
static __device__ __forceinline__ void wait_cp() {
#if USE_ASYNC
    __builtin_amdgcn_s_wait_asynccnt(0);
#endif
}

static __device__ __forceinline__ unsigned short f32_to_bf16_rne(float f) {
    unsigned int u = __float_as_uint(f);
    u += 0x7FFFu + ((u >> 16) & 1u);
    return (unsigned short)(u >> 16);
}

// ---------------- fp32 -> bf16 conversion pass ----------------
__global__ __launch_bounds__(256)
void cvt_kernel(const float* __restrict__ in, unsigned short* __restrict__ out, long n) {
    long i = ((long)blockIdx.x * blockDim.x + threadIdx.x) * 4;
    if (i + 3 < n) {
        float4 v = *(const float4*)(in + i);
        unsigned long long p =
            (unsigned long long)f32_to_bf16_rne(v.x)
          | ((unsigned long long)f32_to_bf16_rne(v.y) << 16)
          | ((unsigned long long)f32_to_bf16_rne(v.z) << 32)
          | ((unsigned long long)f32_to_bf16_rne(v.w) << 48);
        *(unsigned long long*)(out + i) = p;
    } else {
        for (; i < n; ++i) out[i] = f32_to_bf16_rne(in[i]);
    }
}

static __device__ __forceinline__ void expert_range(const int* __restrict__ counts,
                                                    int e, int& off, int& cnt) {
    off = 0; cnt = 0;
#pragma unroll
    for (int i = 0; i < E_; ++i) {
        int c = counts[i];
        if (i < e) off += c;
        if (i == e) cnt = c;
    }
}

// A fragment: lane holds row m=lane&15; K = kk+16h .. kk+16h+15 (contiguous)
static __device__ __forceinline__ v16bf load_a_frag(const __bf16* p) {
    v8bf lo = *(const v8bf*)p;
    v8bf hi = *(const v8bf*)(p + 8);
    return __builtin_shufflevector(lo, hi, 0,1,2,3,4,5,6,7,8,9,10,11,12,13,14,15);
}
// B fragment: lane holds col n=lane&15; K = kk+16h .. kk+16h+15 (contiguous)
static __device__ __forceinline__ v16bf load_b_frag(const __bf16* p) {
    v8bf lo = *(const v8bf*)p;
    v8bf hi = *(const v8bf*)(p + 8);
    return __builtin_shufflevector(lo, hi, 0,1,2,3,4,5,6,7,8,9,10,11,12,13,14,15);
}

// ---------------- GEMM1: h = silu(x W1^T) * (x W3^T)  [bf16 out] ----------------
__global__ __launch_bounds__(256)
void gemm1_kernel(const __bf16* __restrict__ xb,
                  const __bf16* __restrict__ w1b,
                  const __bf16* __restrict__ w3b,
                  const int* __restrict__ counts,
                  __bf16* __restrict__ hb) {
    __shared__ __bf16 sA[2][BM][BKP];
    __shared__ __bf16 sB1[2][BN][BKP];
    __shared__ __bf16 sB3[2][BN][BKP];

    const int e = blockIdx.z;
    int off, cnt;
    expert_range(counts, e, off, cnt);
    const int mbase = blockIdx.y * BM;
    if (mbase >= cnt) return;
    const int nbase = blockIdx.x * BN;

    const int tid  = threadIdx.x;
    const int lane = tid & 31;
    const int n16  = lane & 15;
    const int hl   = lane >> 4;
    const int wave = tid >> 5;
    const int wm   = wave & 3;   // 4 waves along M
    const int wn   = wave >> 2;  // 2 waves along N

    v8f zero = {};
    v8f accg[2][2], accu[2][2];
#pragma unroll
    for (int a = 0; a < 2; ++a)
#pragma unroll
        for (int b = 0; b < 2; ++b) { accg[a][b] = zero; accu[a][b] = zero; }

    const int arow = tid >> 1, ach = tid & 1;   // 2 threads per A row, 32 bf16 each
    const int brow = tid >> 2, bcq = tid & 3;   // 4 threads per B row, 16 bf16 each
    const int agrow = mbase + arow;
    const bool avalid = (agrow < cnt);
    const __bf16* aSrc  = xb  + (size_t)(off + agrow) * D_ + ach * 32;
    const __bf16* b1Src = w1b + ((size_t)e * H_ + (nbase + brow)) * D_ + bcq * 16;
    const __bf16* b3Src = w3b + ((size_t)e * H_ + (nbase + brow)) * D_ + bcq * 16;

    // one-time zero fill for padded tail rows (both buffers), before first barrier
    if (!avalid) {
#pragma unroll
        for (int b = 0; b < 2; ++b) {
            __bf16* da = &sA[b][arow][ach * 32];
#pragma unroll
            for (int j = 0; j < 32; ++j) da[j] = (__bf16)0.0f;
        }
    }

    auto issue = [&](int buf, int k0) {
        if (avalid) {
#pragma unroll
            for (int j = 0; j < 4; ++j)
                cp16(aSrc + k0 + j * 8, &sA[buf][arow][ach * 32 + j * 8]);
        }
#pragma unroll
        for (int j = 0; j < 2; ++j) {
            cp16(b1Src + k0 + j * 8, &sB1[buf][brow][bcq * 16 + j * 8]);
            cp16(b3Src + k0 + j * 8, &sB3[buf][brow][bcq * 16 + j * 8]);
        }
    };

    issue(0, 0);
    for (int k0 = 0, it = 0; k0 < D_; k0 += BK, ++it) {
        const int cur = it & 1;
        wait_cp();
        __syncthreads();
        if (k0 + BK < D_) issue(cur ^ 1, k0 + BK);
#pragma unroll
        for (int kk = 0; kk < BK; kk += 32) {
            v16bf af[2], b1f[2], b3f[2];
#pragma unroll
            for (int tm = 0; tm < 2; ++tm)
                af[tm] = load_a_frag(&sA[cur][wm * 32 + tm * 16 + n16][kk + hl * 16]);
#pragma unroll
            for (int tn = 0; tn < 2; ++tn) {
                b1f[tn] = load_b_frag(&sB1[cur][wn * 32 + tn * 16 + n16][kk + hl * 16]);
                b3f[tn] = load_b_frag(&sB3[cur][wn * 32 + tn * 16 + n16][kk + hl * 16]);
            }
#pragma unroll
            for (int tm = 0; tm < 2; ++tm)
#pragma unroll
                for (int tn = 0; tn < 2; ++tn) {
                    accg[tm][tn] = __builtin_amdgcn_wmma_f32_16x16x32_bf16(
                        false, af[tm], false, b1f[tn], (short)0, accg[tm][tn], false, false);
                    accu[tm][tn] = __builtin_amdgcn_wmma_f32_16x16x32_bf16(
                        false, af[tm], false, b3f[tn], (short)0, accu[tm][tn], false, false);
                }
        }
    }

#pragma unroll
    for (int tm = 0; tm < 2; ++tm)
#pragma unroll
        for (int tn = 0; tn < 2; ++tn) {
            const int col = nbase + wn * 32 + tn * 16 + n16;
#pragma unroll
            for (int r = 0; r < 8; ++r) {
                const int m = mbase + wm * 32 + tm * 16 + hl * 8 + r;
                if (m < cnt) {
                    float g = accg[tm][tn][r];
                    float u = accu[tm][tn][r];
                    float s = g / (1.0f + __expf(-g));   // silu(g)
                    hb[(size_t)(off + m) * H_ + col] = (__bf16)(s * u);
                }
            }
        }
}

// ---------------- GEMM2: y = h W2^T  [fp32 out] ----------------
__global__ __launch_bounds__(256)
void gemm2_kernel(const __bf16* __restrict__ hb,
                  const __bf16* __restrict__ w2b,
                  const int* __restrict__ counts,
                  float* __restrict__ out) {
    __shared__ __bf16 sA[2][BM][BKP];
    __shared__ __bf16 sB[2][BN][BKP];

    const int e = blockIdx.z;
    int off, cnt;
    expert_range(counts, e, off, cnt);
    const int mbase = blockIdx.y * BM;
    if (mbase >= cnt) return;
    const int nbase = blockIdx.x * BN;

    const int tid  = threadIdx.x;
    const int lane = tid & 31;
    const int n16  = lane & 15;
    const int hl   = lane >> 4;
    const int wave = tid >> 5;
    const int wm   = wave & 3;
    const int wn   = wave >> 2;

    v8f zero = {};
    v8f acc[2][2];
#pragma unroll
    for (int a = 0; a < 2; ++a)
#pragma unroll
        for (int b = 0; b < 2; ++b) acc[a][b] = zero;

    const int arow = tid >> 1, ach = tid & 1;
    const int brow = tid >> 2, bcq = tid & 3;
    const int agrow = mbase + arow;
    const bool avalid = (agrow < cnt);
    const __bf16* aSrc = hb  + (size_t)(off + agrow) * H_ + ach * 32;
    const __bf16* bSrc = w2b + ((size_t)e * D_ + (nbase + brow)) * H_ + bcq * 16;

    if (!avalid) {
#pragma unroll
        for (int b = 0; b < 2; ++b) {
            __bf16* da = &sA[b][arow][ach * 32];
#pragma unroll
            for (int j = 0; j < 32; ++j) da[j] = (__bf16)0.0f;
        }
    }

    auto issue = [&](int buf, int k0) {
        if (avalid) {
#pragma unroll
            for (int j = 0; j < 4; ++j)
                cp16(aSrc + k0 + j * 8, &sA[buf][arow][ach * 32 + j * 8]);
        }
#pragma unroll
        for (int j = 0; j < 2; ++j)
            cp16(bSrc + k0 + j * 8, &sB[buf][brow][bcq * 16 + j * 8]);
    };

    issue(0, 0);
    for (int k0 = 0, it = 0; k0 < H_; k0 += BK, ++it) {
        const int cur = it & 1;
        wait_cp();
        __syncthreads();
        if (k0 + BK < H_) issue(cur ^ 1, k0 + BK);
#pragma unroll
        for (int kk = 0; kk < BK; kk += 32) {
            v16bf af[2], bf[2];
#pragma unroll
            for (int tm = 0; tm < 2; ++tm)
                af[tm] = load_a_frag(&sA[cur][wm * 32 + tm * 16 + n16][kk + hl * 16]);
#pragma unroll
            for (int tn = 0; tn < 2; ++tn)
                bf[tn] = load_b_frag(&sB[cur][wn * 32 + tn * 16 + n16][kk + hl * 16]);
#pragma unroll
            for (int tm = 0; tm < 2; ++tm)
#pragma unroll
                for (int tn = 0; tn < 2; ++tn)
                    acc[tm][tn] = __builtin_amdgcn_wmma_f32_16x16x32_bf16(
                        false, af[tm], false, bf[tn], (short)0, acc[tm][tn], false, false);
        }
    }

#pragma unroll
    for (int tm = 0; tm < 2; ++tm)
#pragma unroll
        for (int tn = 0; tn < 2; ++tn) {
            const int col = nbase + wn * 32 + tn * 16 + n16;
#pragma unroll
            for (int r = 0; r < 8; ++r) {
                const int m = mbase + wm * 32 + tm * 16 + hl * 8 + r;
                if (m < cnt)
                    out[(size_t)(off + m) * D_ + col] = acc[tm][tn][r];
            }
        }
}

extern "C" void kernel_launch(void* const* d_in, const int* in_sizes, int n_in,
                              void* d_out, int out_size, void* d_ws, size_t ws_size,
                              hipStream_t stream) {
    const float* w1 = (const float*)d_in[0];
    const float* w2 = (const float*)d_in[1];
    const float* w3 = (const float*)d_in[2];
    const float* x  = (const float*)d_in[3];
    const int* counts = (const int*)d_in[4];

    const long Wn = (long)E_ * H_ * D_;
    const int  T  = in_sizes[3] / D_;
    const long Xn = (long)T * D_;

    // workspace layout (bf16 stored as u16): w1 | w3 | w2 | x | h
    unsigned short* w1b = (unsigned short*)d_ws;
    unsigned short* w3b = w1b + Wn;
    unsigned short* w2b = w3b + Wn;
    unsigned short* xb  = w2b + Wn;
    unsigned short* hb  = xb + Xn;

    {
        int bw = (int)((Wn / 4 + 255) / 256);
        cvt_kernel<<<bw, 256, 0, stream>>>(w1, w1b, Wn);
        cvt_kernel<<<bw, 256, 0, stream>>>(w3, w3b, Wn);
        cvt_kernel<<<bw, 256, 0, stream>>>(w2, w2b, Wn);
        int bx = (int)((Xn / 4 + 255) / 256);
        cvt_kernel<<<bx, 256, 0, stream>>>(x, xb, Xn);
    }

    dim3 blk(256, 1, 1);
    dim3 g1(H_ / BN, (T + BM - 1) / BM, E_);
    gemm1_kernel<<<g1, blk, 0, stream>>>((const __bf16*)xb, (const __bf16*)w1b,
                                         (const __bf16*)w3b, counts, (__bf16*)hb);
    dim3 g2(D_ / BN, (T + BM - 1) / BM, E_);
    gemm2_kernel<<<g2, blk, 0, stream>>>((const __bf16*)hb, (const __bf16*)w2b,
                                         counts, (float*)d_out);
}
